// RAMRecurrentNetwork_44066364457564
// MI455X (gfx1250) — 8-bit matrix operand order repaired and checked by reading, compiled, not run on hardware
//
#include <hip/hip_runtime.h>
#include <stdint.h>

#define INPUT_BITS 1024
#define N_STATE    2048
#define N_OUT      256
#define T_STEPS    64
#define HASHSZ     65536

typedef int v8i __attribute__((ext_vector_type(8)));

#define OFF_WLO     ((size_t)0)
#define OFF_WHI     ((size_t)(2u << 20))
#define OFF_PACKED  ((size_t)(4u << 20))
#define OFF_WINCLO  (OFF_PACKED + ((size_t)64 << 10))
#define OFF_WINCHI  (OFF_WINCLO + ((size_t)512 << 10))
#define OFF_STLIST  (OFF_WINCHI + ((size_t)512 << 10))

__global__ void zero_ws_kernel(uint32_t* p, int n) {
    int i = blockIdx.x * blockDim.x + threadIdx.x;
    if (i < n) p[i] = 0u;
}

__global__ void pack_bits_kernel(const int* __restrict__ in, uint8_t* __restrict__ out, int n) {
    int i = blockIdx.x * blockDim.x + threadIdx.x;
    if (i < n) out[i] = (uint8_t)in[i];
}

__global__ void build_conn_kernel(const int* __restrict__ conn_state,
                                  uint8_t* __restrict__ Wlo,
                                  uint8_t* __restrict__ Whi,
                                  uint32_t* __restrict__ stlist) {
    int n = blockIdx.x * blockDim.x + threadIdx.x;
    if (n >= N_STATE) return;
    int cnt = 0;
    for (int i = 0; i < 16; ++i) {
        int j = conn_state[n * 16 + i];
        if (j < INPUT_BITS) {
            size_t o = (size_t)n * INPUT_BITS + (size_t)j;
            if (i < 8) Wlo[o] = (uint8_t)(Wlo[o] + (uint8_t)(1u << i));
            else       Whi[o] = (uint8_t)(Whi[o] + (uint8_t)(1u << (i - 8)));
        } else {
            stlist[n * 16 + cnt] = ((uint32_t)(j - INPUT_BITS) << 16) | (uint32_t)(1u << i);
            ++cnt;
        }
    }
    for (; cnt < 16; ++cnt) stlist[n * 16 + cnt] = 0u;
}

__global__ __launch_bounds__(32) void win_gemm_kernel(const uint8_t* __restrict__ Wlo,
                                                      const uint8_t* __restrict__ Whi,
                                                      const uint8_t* __restrict__ packed,
                                                      int* __restrict__ winclo,
                                                      int* __restrict__ winchi) {
    int tile  = blockIdx.x;
    int which = tile >> 9;
    int rem   = tile & 511;
    int mt    = rem >> 2;
    int tt    = rem & 3;

    const uint8_t* W    = which ? Whi    : Wlo;
    int*           winc = which ? winchi : winclo;

    int lane = threadIdx.x & 31;
    int m    = lane & 15;
    int h    = lane >> 4;

    const uint8_t* arow = W      + (size_t)(mt * 16 + m) * INPUT_BITS;
    const uint8_t* bcol = packed + (size_t)(tt * 16 + m) * INPUT_BITS;

    v8i acc = {};
    for (int kc = 0; kc < 16; ++kc) {
        v8i a, b;
#pragma unroll
        for (int v = 0; v < 8; ++v) {
            int ka = (v >> 2) * 32 + h * 16 + (v & 3) * 4;
            int kb = (v >> 2) * 32 + h * 16 + (v & 3) * 4;
            a[v] = *(const int*)(arow + kc * 64 + ka);
            b[v] = *(const int*)(bcol + kc * 64 + kb);
        }
        acc = __builtin_amdgcn_wmma_i32_16x16x64_iu8(false, a, false, b, acc, false, false);
    }

#pragma unroll
    for (int r = 0; r < 8; ++r) {
        int t      = tt * 16 + m;
        int neuron = mt * 16 + r + 8 * h;
        winc[t * N_STATE + neuron] = acc[r];
    }
}

__global__ __launch_bounds__(1024) void recurrent_kernel(const int* __restrict__ winclo,
                                                         const int* __restrict__ winchi,
                                                         const uint32_t* __restrict__ stlist,
                                                         const float* __restrict__ state_table,
                                                         const int* __restrict__ conn_out,
                                                         const float* __restrict__ output_table,
                                                         float* __restrict__ out) {
    __shared__ uint8_t st[2][N_STATE];

    int tid = threadIdx.x;
    int n0 = tid, n1 = tid + 1024;

    uint32_t l0[16], l1[16];
#pragma unroll
    for (int i = 0; i < 16; ++i) {
        l0[i] = stlist[n0 * 16 + i];
        l1[i] = stlist[n1 * 16 + i];
    }
    st[0][n0] = 0;
    st[0][n1] = 0;
    __syncthreads();

    int p = 0;
    for (int t = 0; t < T_STEPS; ++t) {
        int a0 = winclo[t * N_STATE + n0] + (winchi[t * N_STATE + n0] << 8);
        int a1 = winclo[t * N_STATE + n1] + (winchi[t * N_STATE + n1] << 8);
#pragma unroll
        for (int i = 0; i < 16; ++i) {
            uint32_t e0 = l0[i], e1 = l1[i];
            a0 += (int)st[p][e0 >> 16] * (int)(e0 & 0xFFFFu);
            a1 += (int)st[p][e1 >> 16] * (int)(e1 & 0xFFFFu);
        }
        float v0 = state_table[(size_t)n0 * HASHSZ + (size_t)a0];
        float v1 = state_table[(size_t)n1 * HASHSZ + (size_t)a1];
        st[p ^ 1][n0] = (v0 > 0.5f) ? 1 : 0;
        st[p ^ 1][n1] = (v1 > 0.5f) ? 1 : 0;
        p ^= 1;
        __syncthreads();
    }

    if (tid < N_OUT) {
        int addr = 0;
#pragma unroll
        for (int i = 0; i < 16; ++i)
            addr += (int)st[p][conn_out[tid * 16 + i]] << i;
        out[tid] = output_table[(size_t)tid * HASHSZ + (size_t)addr];
    }
}

extern "C" void kernel_launch(void* const* d_in, const int* in_sizes, int n_in,
                              void* d_out, int out_size, void* d_ws, size_t ws_size,
                              hipStream_t stream) {
    (void)in_sizes; (void)n_in; (void)out_size; (void)ws_size;

    const int*   input_bits   = (const int*)d_in[0];
    const int*   conn_state   = (const int*)d_in[1];
    const int*   conn_out     = (const int*)d_in[2];
    const float* state_table  = (const float*)d_in[3];
    const float* output_table = (const float*)d_in[4];

    char* ws = (char*)d_ws;
    uint8_t*  Wlo    = (uint8_t*)(ws + OFF_WLO);
    uint8_t*  Whi    = (uint8_t*)(ws + OFF_WHI);
    uint8_t*  packed = (uint8_t*)(ws + OFF_PACKED);
    int*      winclo = (int*)(ws + OFF_WINCLO);
    int*      winchi = (int*)(ws + OFF_WINCHI);
    uint32_t* stlist = (uint32_t*)(ws + OFF_STLIST);

    zero_ws_kernel<<<(1 << 20) / 256, 256, 0, stream>>>((uint32_t*)(ws + OFF_WLO), 1 << 20);
    pack_bits_kernel<<<(T_STEPS * INPUT_BITS) / 256, 256, 0, stream>>>(
        input_bits, packed, T_STEPS * INPUT_BITS);
    build_conn_kernel<<<N_STATE / 256, 256, 0, stream>>>(conn_state, Wlo, Whi, stlist);
    win_gemm_kernel<<<1024, 32, 0, stream>>>(Wlo, Whi, packed, winclo, winchi);
    recurrent_kernel<<<1, 1024, 0, stream>>>(winclo, winchi, stlist, state_table,
                                             conn_out, output_table, (float*)d_out);
}
